// RGBMambaBranch_39393440039559
// MI455X (gfx1250) — compile-verified, matrix operand order checked by reading, not run on hardware
//
#include <hip/hip_runtime.h>
#include <hip/hip_bf16.h>
#include <cstdint>
#include <cstddef>

// ---------------------------------------------------------------------------
// RGBMambaBranch for MI455X (gfx1250, wave32, WMMA).
// Roofline: ~127 GFLOP total, dominated by the stem convs (~93 GFLOP conv1,
// ~14 GFLOP conv2, ~19 GFLOP conv3d).  Input traffic 126MB -> ~5.4us at
// 23.3 TB/s, so stem convs are compute bound => implicit-GEMM + v_wmma f16.
// Block stack (24 x ~60 MFLOP) is latency bound; FFT(L=80) is lowered to
// 80x80 DFT GEMMs so it also rides the WMMA path.
// Fragment-major B tiles in LDS (ds_load_b128 fragments) and ASYNCcnt-tracked
// GLOBAL_LOAD_ASYNC_TO_LDS_B128 staging for GEMM tiles / conv3d weights.
// ---------------------------------------------------------------------------

#define BN_SCALE_F 0.9999950000374997f   // 1/sqrt(1+1e-5)
#define C_ED   96
#define C_SD   24
#define C_DI   192
#define C_NST  16
#define C_L    80
#define C_B    4
#define C_D    160
#define C_BL   (C_B * C_L)               // 320 tokens

typedef __attribute__((ext_vector_type(16))) _Float16 v16h;
typedef __attribute__((ext_vector_type(8)))  float    v8f;
typedef __attribute__((ext_vector_type(4)))  int      v4i;

__device__ __forceinline__ int imin(int a, int b) { return a < b ? a : b; }
__device__ __forceinline__ int imax(int a, int b) { return a > b ? a : b; }
__device__ __forceinline__ float sigm(float v) { return 1.f / (1.f + expf(-v)); }

// --- CDNA5 async global->LDS copy (ASYNCcnt), with safe fallback -----------
#if defined(__HIP_DEVICE_COMPILE__) && \
    __has_builtin(__builtin_amdgcn_global_load_async_to_lds_b128) && \
    __has_builtin(__builtin_amdgcn_s_wait_asynccnt)
#define HAVE_ASYNC_LDS 1
#else
#define HAVE_ASYNC_LDS 0
#endif

// builtin signature (from hipcc diagnostic): first param is
// "int __vector(4) __device__ *" i.e. v4i* in the global AS; LDS dest is the
// matching v4i* in the shared AS; trailing (imm offset, imm cpol).
typedef __attribute__((address_space(1))) v4i gas_v4i;
typedef __attribute__((address_space(3))) v4i las_v4i;

__device__ __forceinline__ void async_copy16(const void* g, void* l)
{
#if HAVE_ASYNC_LDS
    __builtin_amdgcn_global_load_async_to_lds_b128((gas_v4i*)g, (las_v4i*)l, 0, 0);
#else
    (void)g; (void)l;
#endif
}
__device__ __forceinline__ void async_wait0()
{
#if HAVE_ASYNC_LDS
    __builtin_amdgcn_s_wait_asynccnt(0);
#endif
}

// ===========================================================================
// Generic WMMA GEMM:  C[bz] = act(alpha*A[bz]·B[bz]^T + bias) + res + beta*C
//   A: [M][K] row-major (lda), batch stride sAb (elements)
//   B: transB==0 -> [N][K] (weights); transB==1 -> [K][N]
//   act: 0 none, 1 relu, 2 softplus
// Tile: 256 threads = 8 waves, 64x64 C tile, K-step 32.
// Fast path: GLOBAL_LOAD_ASYNC_TO_LDS_B128 raw-f32 staging + one LDS f32->f16
// convert pass; fragments then load as 2x ds_load_b128 per operand.
// ===========================================================================
__global__ __launch_bounds__(256) void k_gemm_nt(
    const float* __restrict__ A, long long sAb, int lda,
    const float* __restrict__ B, long long sBb, int ldb, int transB,
    float*       __restrict__ C, long long sCb, int ldc,
    const float* __restrict__ bias,
    const float* __restrict__ res, long long sRb,
    int M, int N, int K, float alpha, float beta, int act)
{
    __shared__ _Float16 As [64 * 32];   // [m][k]
    __shared__ _Float16 Bs [64 * 32];   // [n][k]  (fragment-major)
    __shared__ float    Asf[64 * 32];   // async raw staging [m][k]
    __shared__ float    Bsf[64 * 32];   // async raw: transB? [k][n] : [n][k]

    constexpr bool kAsync = (HAVE_ASYNC_LDS != 0);

    const int bz    = blockIdx.z;
    const int mBase = blockIdx.x * 64;
    const int nBase = blockIdx.y * 64;
    A += (long long)bz * sAb;
    B += (long long)bz * sBb;
    C += (long long)bz * sCb;

    const int tid    = threadIdx.x;
    const int wave   = tid >> 5;
    const int lane   = tid & 31;
    const int mt     = wave & 3;           // 4 M-tiles of 16
    const int ntBase = (wave >> 2) * 2;    // waves 0-3 -> nt {0,1}; 4-7 -> {2,3}
    const int kShift = (lane < 16) ? 0 : 8;
    const int nn     = lane & 15;
    const int am     = (mt * 16 + nn) * 32;
    const int bn0    = (ntBase * 16 + nn) * 32;
    const int bn1    = (ntBase * 16 + 16 + nn) * 32;

    v8f acc0 = {};
    v8f acc1 = {};

    for (int k0 = 0; k0 < K; k0 += 32) {
        const bool kFull = (k0 + 32 <= K);
        const bool aFast = kAsync && kFull && (mBase + 64 <= M) && ((lda & 3) == 0);
        const bool bFast = kAsync && kFull && (nBase + 64 <= N) && ((ldb & 3) == 0);

        if (aFast) {
            // 64 rows x 128B = 512 16B chunks, 2 per thread
            for (int c = tid; c < 512; c += 256) {
                int m = c >> 3, part = c & 7;
                async_copy16(A + (long long)(mBase + m) * lda + k0 + part * 4,
                             Asf + c * 4);
            }
        } else {
            for (int i = tid; i < 64 * 32; i += 256) {
                int m = i >> 5, k = i & 31;
                int gm = mBase + m, gk = k0 + k;
                As[i] = (gm < M && gk < K) ? (_Float16)A[(long long)gm * lda + gk]
                                           : (_Float16)0.f;
            }
        }
        if (bFast) {
            if (transB) {          // rows of 64 contiguous n per k
                for (int c = tid; c < 512; c += 256) {
                    int k = c >> 4, part = c & 15;
                    async_copy16(B + (long long)(k0 + k) * ldb + nBase + part * 4,
                                 Bsf + c * 4);
                }
            } else {               // rows of 32 contiguous k per n
                for (int c = tid; c < 512; c += 256) {
                    int n = c >> 3, part = c & 7;
                    async_copy16(B + (long long)(nBase + n) * ldb + k0 + part * 4,
                                 Bsf + c * 4);
                }
            }
        } else {
            for (int i = tid; i < 64 * 32; i += 256) {
                int n = i >> 5, k = i & 31;
                int gn = nBase + n, gk = k0 + k;
                float v = 0.f;
                if (gn < N && gk < K)
                    v = transB ? B[(long long)gk * ldb + gn]
                               : B[(long long)gn * ldb + gk];
                Bs[i] = (_Float16)v;
            }
        }
        if (aFast || bFast) async_wait0();   // per-wave ASYNCcnt drain
        __syncthreads();
        if (aFast)
            for (int i = tid; i < 64 * 32; i += 256) As[i] = (_Float16)Asf[i];
        if (bFast) {
            if (transB)
                for (int i = tid; i < 64 * 32; i += 256) {
                    int n = i >> 5, k = i & 31;
                    Bs[i] = (_Float16)Bsf[k * 64 + n];
                }
            else
                for (int i = tid; i < 64 * 32; i += 256) Bs[i] = (_Float16)Bsf[i];
        }
        if (aFast || bFast) __syncthreads();

        // fragments: lanes 0-15 carry K {0..7,16..23}; 16-31 {8..15,24..31};
        // both operands are contiguous 8-half runs -> ds_load_b128 pairs
        v16h a, b0, b1;
        #pragma unroll
        for (int i = 0; i < 16; ++i) {
            int k = kShift + ((i < 8) ? i : (i + 8));
            a[i]  = As[am + k];
            b0[i] = Bs[bn0 + k];
            b1[i] = Bs[bn1 + k];
        }
        acc0 = __builtin_amdgcn_wmma_f32_16x16x32_f16(false, a, false, b0,
                                                      (short)0, acc0, false, false);
        acc1 = __builtin_amdgcn_wmma_f32_16x16x32_f16(false, a, false, b1,
                                                      (short)0, acc1, false, false);
        __syncthreads();
    }

    // epilogue: C layout VGPR r -> M = r (lanes 0-15) / 8+r (lanes 16-31)
    #pragma unroll
    for (int r = 0; r < 8; ++r) {
        int gm = mBase + mt * 16 + r + ((lane < 16) ? 0 : 8);
        #pragma unroll
        for (int t = 0; t < 2; ++t) {
            int gn = nBase + (ntBase + t) * 16 + nn;
            if (gm < M && gn < N) {
                float v = alpha * (t ? acc1[r] : acc0[r]);
                if (bias) v += bias[gn];
                if (act == 1)      v = fmaxf(v, 0.f);
                else if (act == 2) v = (v > 20.f) ? v : log1pf(expf(v));
                if (res)  v += res[(long long)bz * sRb + (long long)gm * ldc + gn];
                if (beta != 0.f) v += beta * C[(long long)gm * ldc + gn];
                C[(long long)gm * ldc + gn] = v;
            }
        }
    }
}

// ===========================================================================
// Implicit-GEMM conv2d (Cout=24) with fused relu(BN_SCALE*(y+b)), f16 out.
// mode 0: input = x frames (g -> (b,d), 3 ch)
// mode 1: input = temporal diff of x (12 ch), built on the fly (never
//         materialized: saves 252MB of HBM traffic)
// mode 2: input = f32 buffer (G,24,Hin,Win)
// 128 threads = 4 waves; WG computes an 8x8 output tile x 24 couts.
// Weights held fragment-major [cout][Kpad] so B frags are b128 LDS loads.
// ===========================================================================
#define CONV_MAXW (32 * 608)   // 32 couts x Kpad(<=608), f16
#define CONV_MAXP (21 * 21 * 12)
__global__ __launch_bounds__(128) void k_conv2d(
    const float* __restrict__ x, const float* __restrict__ src,
    const float* __restrict__ w, const float* __restrict__ bias,
    _Float16* __restrict__ out,
    int mode, int Cin, int KS, int stride, int pad,
    int Hin, int Win, int Hout, int Wout)
{
    __shared__ _Float16 sW[CONV_MAXW];   // [co][k]
    __shared__ _Float16 sP[CONV_MAXP];

    const int tX = Wout >> 3, tY = Hout >> 3;
    const int g  = blockIdx.x / (tX * tY);
    const int t  = blockIdx.x % (tX * tY);
    const int ty = t / tX, tx = t % tX;
    const int K    = Cin * KS * KS;
    const int Kpad = (K + 31) & ~31;
    const int KSS  = KS * KS;
    const int tid  = threadIdx.x;

    // weights -> LDS fragment-major [cout(32 padded)][k]
    for (int co = 0; co < 32; ++co)
        for (int k = tid; k < Kpad; k += 128)
            sW[co * Kpad + k] = (co < 24 && k < K) ? (_Float16)w[co * K + k]
                                                   : (_Float16)0.f;

    // input patch -> LDS (PH = (8-1)*stride + KS), zero padded
    const int PH = 7 * stride + KS;
    const int r0 = ty * 8 * stride - pad;
    const int c0 = tx * 8 * stride - pad;
    const int dIdx = g % C_D, bIm = g / C_D;
    const int npat = Cin * PH * PH;
    for (int i = tid; i < npat; i += 128) {
        int cin = i / (PH * PH);
        int rem = i % (PH * PH);
        int rr = rem / PH, cc = rem % PH;
        int gh = r0 + rr, gw = c0 + cc;
        float v = 0.f;
        if (gh >= 0 && gh < Hin && gw >= 0 && gw < Win) {
            if (mode == 0) {
                v = x[(((long long)g * 3 + cin) * Hin + gh) * Win + gw];
            } else if (mode == 1) {
                int grp = cin / 3, c = cin % 3;
                int fa, fb;
                if (grp == 0)      { fa = imax(dIdx - 1, 0);        fb = imax(dIdx - 2, 0); }
                else if (grp == 1) { fa = dIdx;                     fb = imax(dIdx - 1, 0); }
                else if (grp == 2) { fa = imin(dIdx + 1, C_D - 1);  fb = dIdx; }
                else               { fa = imin(dIdx + 2, C_D - 1);  fb = imin(dIdx + 1, C_D - 1); }
                long long base = (long long)bIm * C_D;
                v = x[(((base + fa) * 3 + c) * Hin + gh) * Win + gw]
                  - x[(((base + fb) * 3 + c) * Hin + gh) * Win + gw];
            } else {
                v = src[(((long long)g * 24 + cin) * Hin + gh) * Win + gw];
            }
        }
        sP[i] = (_Float16)v;
    }
    __syncthreads();

    const int wave = tid >> 5, lane = tid & 31;
    const int mt = wave;
    const int m  = mt * 16 + (lane & 15);
    const int pr = m >> 3, pc = m & 7;
    const int kShift = (lane < 16) ? 0 : 8;
    const int nn = lane & 15;

    v8f acc0 = {}, acc1 = {};
    for (int k0 = 0; k0 < Kpad; k0 += 32) {
        v16h a, b0, b1;
        #pragma unroll
        for (int i = 0; i < 16; ++i) {
            int k = k0 + kShift + ((i < 8) ? i : (i + 8));
            _Float16 av = (_Float16)0.f;
            if (k < K) {
                int cin = k / KSS, rem = k % KSS;
                int kh = rem / KS, kw = rem % KS;
                av = sP[(cin * PH + pr * stride + kh) * PH + pc * stride + kw];
            }
            a[i]  = av;
            b0[i] = sW[nn * Kpad + k];
            b1[i] = sW[(16 + nn) * Kpad + k];
        }
        acc0 = __builtin_amdgcn_wmma_f32_16x16x32_f16(false, a, false, b0,
                                                      (short)0, acc0, false, false);
        acc1 = __builtin_amdgcn_wmma_f32_16x16x32_f16(false, a, false, b1,
                                                      (short)0, acc1, false, false);
    }

    #pragma unroll
    for (int r = 0; r < 8; ++r) {
        int m2 = mt * 16 + r + ((lane < 16) ? 0 : 8);
        int oh = ty * 8 + (m2 >> 3), ow = tx * 8 + (m2 & 7);
        float v0 = fmaxf(BN_SCALE_F * (acc0[r] + bias[nn]), 0.f);
        out[(((long long)g * 24 + nn) * Hout + oh) * Wout + ow] = (_Float16)v0;
        int co1 = 16 + nn;
        if (co1 < 24) {
            float v1 = fmaxf(BN_SCALE_F * (acc1[r] + bias[co1]), 0.f);
            out[(((long long)g * 24 + co1) * Hout + oh) * Wout + ow] = (_Float16)v1;
        }
    }
}

// ===========================================================================
// Implicit-GEMM conv3d: w3 (96,24,2,5,5), stride (2,1,1), pad (0,2,2).
// y5 layout (B,24,160,16,16) f32 -> out (B,96,80,16,16) = BN_SCALE*(conv+b3).
// WG = one (b,dz), one 8x8 spatial tile; N=96 -> 6 N-tiles per wave's M-tile.
// K = 1200 (pad 1216); 32xK weight slice restaged per step, via async LDS
// copy (full slices) -> convert, fragment-major [co][32].
// ===========================================================================
__global__ __launch_bounds__(128) void k_conv3d(
    const float* __restrict__ y5, const float* __restrict__ w3,
    const float* __restrict__ b3, float* __restrict__ out)
{
    __shared__ _Float16 sW [96 * 32];   // [co][kk]
    __shared__ float    sWf[96 * 32];   // async raw staging [co][kk]
    __shared__ _Float16 sP [48 * 144];  // (ci*2+kd) x 12 x 12

    constexpr bool kAsync = (HAVE_ASYNC_LDS != 0);

    const int id   = blockIdx.x;        // B*80*4
    const int tile = id & 3;
    const int dz   = (id >> 2) % C_L;
    const int b    = id / (4 * C_L);
    const int ty = tile >> 1, tx = tile & 1;
    const int tid = threadIdx.x;
    const int r0 = ty * 8 - 2, c0 = tx * 8 - 2;

    for (int i = tid; i < 48 * 144; i += 128) {
        int cd = i / 144, rem = i % 144;
        int rr = rem / 12, cc = rem % 12;
        int ci = cd >> 1, kd = cd & 1;
        int gh = r0 + rr, gw = c0 + cc;
        float v = 0.f;
        if (gh >= 0 && gh < 16 && gw >= 0 && gw < 16)
            v = y5[(((long long)b * 24 + ci) * C_D + (2 * dz + kd)) * 256 + gh * 16 + gw];
        sP[i] = (_Float16)v;
    }

    const int wave = tid >> 5, lane = tid & 31;
    const int mt = wave;
    const int m  = mt * 16 + (lane & 15);
    const int pr = m >> 3, pc = m & 7;
    const int kShift = (lane < 16) ? 0 : 8;
    const int nn = lane & 15;
    const int K = 1200;

    v8f acc[6];
    v8f zero = {};
    #pragma unroll
    for (int j = 0; j < 6; ++j) acc[j] = zero;

    for (int k0 = 0; k0 < 1216; k0 += 32) {
        __syncthreads();                       // also covers initial sP stage
        const bool wFast = kAsync && (k0 + 32 <= K);
        if (wFast) {
            // 96 rows x 128B = 768 16B chunks, 6 per thread
            for (int c = tid; c < 768; c += 128) {
                int co = c >> 3, part = c & 7;
                async_copy16(w3 + (long long)co * K + k0 + part * 4,
                             sWf + c * 4);
            }
            async_wait0();
        } else {
            for (int i = tid; i < 96 * 32; i += 128) {
                int co = i >> 5, kk = i & 31;
                int k = k0 + kk;
                sW[i] = (k < K) ? (_Float16)w3[(long long)co * K + k]
                                : (_Float16)0.f;
            }
        }
        __syncthreads();
        if (wFast) {
            for (int i = tid; i < 96 * 32; i += 128) sW[i] = (_Float16)sWf[i];
            __syncthreads();
        }

        v16h a;
        #pragma unroll
        for (int i = 0; i < 16; ++i) {
            int kk = kShift + ((i < 8) ? i : (i + 8));
            int k  = k0 + kk;
            _Float16 av = (_Float16)0.f;
            if (k < K) {
                int ci = k / 50, rem = k % 50;
                int kd = rem / 25, r2 = rem % 25;
                int kh = r2 / 5, kw = r2 % 5;
                av = sP[(ci * 2 + kd) * 144 + (pr + kh) * 12 + (pc + kw)];
            }
            a[i] = av;
        }
        #pragma unroll
        for (int j = 0; j < 6; ++j) {
            v16h bfrag;
            #pragma unroll
            for (int i = 0; i < 16; ++i) {
                int kk = kShift + ((i < 8) ? i : (i + 8));
                bfrag[i] = sW[(j * 16 + nn) * 32 + kk];
            }
            acc[j] = __builtin_amdgcn_wmma_f32_16x16x32_f16(false, a, false, bfrag,
                                                            (short)0, acc[j], false, false);
        }
    }

    #pragma unroll
    for (int j = 0; j < 6; ++j) {
        #pragma unroll
        for (int r = 0; r < 8; ++r) {
            int m2 = mt * 16 + r + ((lane < 16) ? 0 : 8);
            int oh = ty * 8 + (m2 >> 3), ow = tx * 8 + (m2 & 7);
            int co = j * 16 + nn;
            float v = BN_SCALE_F * (acc[j][r] + b3[co]);
            out[(((long long)b * 96 + co) * C_L + dz) * 256 + oh * 16 + ow] = v;
        }
    }
}

// ===========================================================================
// Pool / elementwise / scan kernels (bandwidth-trivial pieces)
// ===========================================================================
__global__ void k_pool1(const _Float16* __restrict__ a11,
                        const _Float16* __restrict__ a12,
                        float* __restrict__ p_avg, float* __restrict__ p_d)
{
    long long i = (long long)blockIdx.x * blockDim.x + threadIdx.x;
    if (i >= (long long)640 * 24 * 32 * 32) return;
    int ow = i & 31, oh = (i >> 5) & 31;
    long long gc = i >> 10;
    float m1 = -1e30f, m2 = -1e30f;
    for (int dh = -1; dh <= 1; ++dh)
        for (int dw = -1; dw <= 1; ++dw) {
            int ih = oh * 2 + dh, iw = ow * 2 + dw;
            if (ih >= 0 && ih < 64 && iw >= 0 && iw < 64) {
                long long idx = (gc << 12) + ih * 64 + iw;
                m1 = fmaxf(m1, (float)a11[idx]);
                m2 = fmaxf(m2, (float)a12[idx]);
            }
        }
    p_avg[i] = 0.5f * (m1 + m2);
    p_d[i]   = m2;
}

// pool 32x32 -> 16x16 for both branches, combine, write transposed y5 layout
__global__ void k_pool2(const _Float16* __restrict__ c21,
                        const _Float16* __restrict__ c22,
                        float* __restrict__ y5)
{
    long long i = (long long)blockIdx.x * blockDim.x + threadIdx.x;
    if (i >= (long long)640 * 24 * 16 * 16) return;
    int ow = i & 15, oh = (i >> 4) & 15;
    long long gc = i >> 8;
    int c = (int)(gc % 24);
    long long g = gc / 24;
    int d = (int)(g % C_D), b = (int)(g / C_D);
    float m1 = -1e30f, m2 = -1e30f;
    for (int dh = -1; dh <= 1; ++dh)
        for (int dw = -1; dw <= 1; ++dw) {
            int ih = oh * 2 + dh, iw = ow * 2 + dw;
            if (ih >= 0 && ih < 32 && iw >= 0 && iw < 32) {
                long long idx = gc * 1024 + ih * 32 + iw;
                m1 = fmaxf(m1, (float)c21[idx]);
                m2 = fmaxf(m2, (float)c22[idx]);
            }
        }
    y5[(((long long)b * 24 + c) * C_D + d) * 256 + oh * 16 + ow] = 0.5f * (m1 + m2);
}

// sigmoid-gate + normalized spatial mean -> seq tokens h[b][l][e]
__global__ void k_gatepool(const float* __restrict__ yg, float* __restrict__ h)
{
    int i = blockIdx.x * blockDim.x + threadIdx.x;
    if (i >= C_B * C_ED * C_L) return;
    int dz = i % C_L;
    int co = (i / C_L) % C_ED;
    int b  = i / (C_L * C_ED);
    const float* p = yg + (((long long)b * C_ED + co) * C_L + dz) * 256;
    float s1 = 0.f, s2 = 0.f;
    for (int j = 0; j < 256; ++j) {
        float y = p[j];
        float mg = sigm(y);
        s1 += mg;
        s2 += y * mg;
    }
    h[((long long)b * C_L + dz) * C_ED + co] = 0.5f * s2 / s1;
}

// layernorm over last dim (96); one wave32 per row, 3 elems/lane
__global__ __launch_bounds__(32) void k_ln(const float* __restrict__ x,
                                           const float* __restrict__ gam,
                                           const float* __restrict__ bet,
                                           float* __restrict__ o)
{
    int row  = blockIdx.x;
    int lane = threadIdx.x;
    const float* xr = x + (long long)row * C_ED;
    float v0 = xr[lane], v1 = xr[lane + 32], v2 = xr[lane + 64];
    float s = v0 + v1 + v2;
    for (int off = 16; off; off >>= 1) s += __shfl_xor(s, off, 32);
    float mu = s * (1.f / 96.f);
    float d0 = v0 - mu, d1 = v1 - mu, d2 = v2 - mu;
    float q = d0 * d0 + d1 * d1 + d2 * d2;
    for (int off = 16; off; off >>= 1) q += __shfl_xor(q, off, 32);
    float rs = rsqrtf(q * (1.f / 96.f) + 1e-5f);
    float* orow = o + (long long)row * C_ED;
    orow[lane]      = d0 * rs * gam[lane]      + bet[lane];
    orow[lane + 32] = d1 * rs * gam[lane + 32] + bet[lane + 32];
    orow[lane + 64] = d2 * rs * gam[lane + 64] + bet[lane + 64];
}

// depthwise causal conv(k=4) + bias + silu -> u
__global__ void k_dwconv(const float* __restrict__ xz,
                         const float* __restrict__ cw,
                         const float* __restrict__ cb,
                         float* __restrict__ u)
{
    int i = blockIdx.x * blockDim.x + threadIdx.x;
    if (i >= C_BL * C_DI) return;
    int c  = i % C_DI;
    int bl = i / C_DI;
    int b = bl / C_L, l = bl % C_L;
    float acc = cb[c];
    for (int j = 0; j < 4; ++j) {
        int ll = l - 3 + j;
        if (ll >= 0)
            acc += xz[((long long)(b * C_L + ll)) * (2 * C_DI) + c] * cw[c * 4 + j];
    }
    u[(long long)bl * C_DI + c] = acc * sigm(acc);
}

// selective scan: one wave per (b, channel); lanes = 16 SSM states
__global__ __launch_bounds__(32) void k_scan(
    const float* __restrict__ dtb, const float* __restrict__ dbl,
    const float* __restrict__ u, const float* __restrict__ A_log,
    const float* __restrict__ Dp, float* __restrict__ ys)
{
    int id = blockIdx.x;                 // B*DI
    int c = id % C_DI, b = id / C_DI;
    int lane = threadIdx.x;
    int n = lane & 15;
    float An = -expf(A_log[c * C_NST + n]);
    float hst = 0.f;
    for (int l = 0; l < C_L; ++l) {
        long long bl = (long long)b * C_L + l;
        float dt = dtb[bl * C_DI + c];
        float uu = u[bl * C_DI + c];
        float Bm = dbl[bl * 38 + 6 + n];
        float Cm = dbl[bl * 38 + 22 + n];
        hst = expf(dt * An) * hst + dt * Bm * uu;
        float p = hst * Cm;
        for (int off = 8; off; off >>= 1) p += __shfl_xor(p, off, 16);
        if (lane == 0) ys[bl * C_DI + c] = p + uu * Dp[c];
    }
}

// y *= silu(z)  (z = second half of xz)
__global__ void k_gatez(float* __restrict__ ys, const float* __restrict__ xz)
{
    int i = blockIdx.x * blockDim.x + threadIdx.x;
    if (i >= C_BL * C_DI) return;
    int c = i % C_DI, bl = i / C_DI;
    float z = xz[(long long)bl * (2 * C_DI) + C_DI + c];
    ys[i] *= z * sigm(z);
}

// spectral diag mix: relu(fr*dr - fi*di + rb), relu(fi*dr + fr*di + ib)
__global__ void k_specmix(const float* __restrict__ fr, const float* __restrict__ fi,
                          const float* __restrict__ rmat, const float* __restrict__ imat,
                          const float* __restrict__ rb, const float* __restrict__ ib,
                          float* __restrict__ rp, float* __restrict__ ip)
{
    int i = blockIdx.x * blockDim.x + threadIdx.x;
    if (i >= C_BL * C_DI) return;
    int c = i % C_DI;
    float dr = rmat[c * C_DI + c];
    float di = imat[c * C_DI + c];
    float a = fr[i], b2 = fi[i];
    rp[i] = fmaxf(a * dr - b2 * di + rb[c], 0.f);
    ip[i] = fmaxf(b2 * dr + a * di + ib[c], 0.f);
}

// ortho DFT matrices (80x80): forward exp(-i), inverse exp(+i); symmetric in k*l
__global__ void k_dftinit(float* Fr, float* Fi, float* Ir, float* Ii)
{
    int i = blockIdx.x * blockDim.x + threadIdx.x;
    if (i >= C_L * C_L) return;
    int k = i / C_L, l = i % C_L;
    float th = 6.283185307179586f * (float)(k * l) / (float)C_L;
    float s = sinf(th), c2 = cosf(th);
    const float inv = 0.11180339887498949f;   // 1/sqrt(80)
    Fr[i] = c2 * inv;
    Fi[i] = -s * inv;
    Ir[i] = c2 * inv;
    Ii[i] = s * inv;
}

// ===========================================================================
// Host orchestration
// ===========================================================================
extern "C" void kernel_launch(void* const* d_in, const int* in_sizes, int n_in,
                              void* d_out, int out_size, void* d_ws, size_t ws_size,
                              hipStream_t stream)
{
    (void)in_sizes; (void)n_in; (void)out_size; (void)ws_size;

    const float* x       = (const float*)d_in[0];
    const float* w11     = (const float*)d_in[1];
    const float* b11     = (const float*)d_in[2];
    const float* w12     = (const float*)d_in[3];
    const float* b12     = (const float*)d_in[4];
    const float* w21     = (const float*)d_in[5];
    const float* b21     = (const float*)d_in[6];
    const float* w22     = (const float*)d_in[7];
    const float* b22     = (const float*)d_in[8];
    const float* w3      = (const float*)d_in[9];
    const float* b3      = (const float*)d_in[10];
    const float* ln1_g   = (const float*)d_in[11];
    const float* ln1_b   = (const float*)d_in[12];
    const float* ln2_g   = (const float*)d_in[13];
    const float* ln2_b   = (const float*)d_in[14];
    const float* in_proj = (const float*)d_in[15];
    const float* conv_w  = (const float*)d_in[16];
    const float* conv_b  = (const float*)d_in[17];
    const float* x_proj  = (const float*)d_in[18];
    const float* dt_w    = (const float*)d_in[19];
    const float* dt_b    = (const float*)d_in[20];
    const float* A_log   = (const float*)d_in[21];
    const float* Dp      = (const float*)d_in[22];
    const float* out_p   = (const float*)d_in[23];
    const float* fc1     = (const float*)d_in[24];
    const float* rmat    = (const float*)d_in[25];
    const float* imat    = (const float*)d_in[26];
    const float* rb      = (const float*)d_in[27];
    const float* ib      = (const float*)d_in[28];
    const float* fc2     = (const float*)d_in[29];

    // --- linear workspace allocator (needs ~492MB; assumed available) ---
    char* W = (char*)d_ws;
    size_t cur = 0;
    auto alloc = [&](size_t bytes) { char* p = W + cur; cur += (bytes + 255) & ~(size_t)255; return p; };

    _Float16* a11   = (_Float16*)alloc((size_t)640 * 24 * 64 * 64 * 2);
    _Float16* a12   = (_Float16*)alloc((size_t)640 * 24 * 64 * 64 * 2);
    float*    p_avg = (float*)   alloc((size_t)640 * 24 * 32 * 32 * 4);
    float*    p_d   = (float*)   alloc((size_t)640 * 24 * 32 * 32 * 4);
    _Float16* c21   = (_Float16*)alloc((size_t)640 * 24 * 32 * 32 * 2);
    _Float16* c22   = (_Float16*)alloc((size_t)640 * 24 * 32 * 32 * 2);
    float*    y5    = (float*)   alloc((size_t)C_B * 24 * C_D * 256 * 4);
    float*    yg    = (float*)   alloc((size_t)C_B * 96 * C_L * 256 * 4);
    float*    h     = (float*)   alloc((size_t)C_BL * C_ED * 4);
    float*    xn    = (float*)   alloc((size_t)C_BL * C_ED * 4);
    float*    xz    = (float*)   alloc((size_t)C_BL * 2 * C_DI * 4);
    float*    u     = (float*)   alloc((size_t)C_BL * C_DI * 4);
    float*    dbl   = (float*)   alloc((size_t)C_BL * 38 * 4);
    float*    dtb   = (float*)   alloc((size_t)C_BL * C_DI * 4);
    float*    ys    = (float*)   alloc((size_t)C_BL * C_DI * 4);
    float*    hm    = (float*)   alloc((size_t)C_BL * C_DI * 4);
    float*    fr    = (float*)   alloc((size_t)C_BL * C_DI * 4);
    float*    fi    = (float*)   alloc((size_t)C_BL * C_DI * 4);
    float*    rp    = (float*)   alloc((size_t)C_BL * C_DI * 4);
    float*    ip    = (float*)   alloc((size_t)C_BL * C_DI * 4);
    float*    gbuf  = (float*)   alloc((size_t)C_BL * C_DI * 4);
    float*    dftFr = (float*)   alloc((size_t)C_L * C_L * 4);
    float*    dftFi = (float*)   alloc((size_t)C_L * C_L * 4);
    float*    dftIr = (float*)   alloc((size_t)C_L * C_L * 4);
    float*    dftIi = (float*)   alloc((size_t)C_L * C_L * 4);

    auto gemm = [&](const float* A, long long sAb, int lda,
                    const float* B, long long sBb, int ldb, int transB,
                    float* C, long long sCb, int ldc,
                    const float* bias, const float* res, long long sRb,
                    int M, int N, int K, float alpha, float beta, int act, int batch) {
        dim3 grid((M + 63) / 64, (N + 63) / 64, batch);
        k_gemm_nt<<<grid, 256, 0, stream>>>(A, sAb, lda, B, sBb, ldb, transB,
                                            C, sCb, ldc, bias, res, sRb,
                                            M, N, K, alpha, beta, act);
    };

    // ------------------- stem -------------------
    k_conv2d<<<640 * 64, 128, 0, stream>>>(x, nullptr, w11, b11, a11,
                                           0, 3, 7, 2, 3, 128, 128, 64, 64);
    k_conv2d<<<640 * 64, 128, 0, stream>>>(x, nullptr, w12, b12, a12,
                                           1, 12, 7, 2, 3, 128, 128, 64, 64);
    k_pool1<<<(640 * 24 * 1024 + 255) / 256, 256, 0, stream>>>(a11, a12, p_avg, p_d);
    k_conv2d<<<640 * 16, 128, 0, stream>>>(nullptr, p_avg, w21, b21, c21,
                                           2, 24, 3, 1, 1, 32, 32, 32, 32);
    k_conv2d<<<640 * 16, 128, 0, stream>>>(nullptr, p_d, w22, b22, c22,
                                           2, 24, 3, 1, 1, 32, 32, 32, 32);
    k_pool2<<<(640 * 24 * 256 + 255) / 256, 256, 0, stream>>>(c21, c22, y5);
    k_conv3d<<<C_B * C_L * 4, 128, 0, stream>>>(y5, w3, b3, yg);
    k_gatepool<<<(C_B * C_ED * C_L + 255) / 256, 256, 0, stream>>>(yg, h);
    k_dftinit<<<(C_L * C_L + 255) / 256, 256, 0, stream>>>(dftFr, dftFi, dftIr, dftIi);

    // ------------------- 24 Mamba + FFT-FFN blocks -------------------
    const int nEW = (C_BL * C_DI + 255) / 256;   // 240 blocks for 61440 elems
    for (int blk = 0; blk < 24; ++blk) {
        const float* bl1g = ln1_g + blk * C_ED;
        const float* bl1b = ln1_b + blk * C_ED;
        const float* bl2g = ln2_g + blk * C_ED;
        const float* bl2b = ln2_b + blk * C_ED;
        const float* bip  = in_proj + (long long)blk * 2 * C_DI * C_ED;
        const float* bcw  = conv_w + blk * C_DI * 4;
        const float* bcb  = conv_b + blk * C_DI;
        const float* bxp  = x_proj + (long long)blk * 38 * C_DI;
        const float* bdtw = dt_w + blk * C_DI * 6;
        const float* bdtb = dt_b + blk * C_DI;
        const float* bAl  = A_log + blk * C_DI * C_NST;
        const float* bDp  = Dp + blk * C_DI;
        const float* bop  = out_p + (long long)blk * C_ED * C_DI;
        const float* bf1  = fc1 + (long long)blk * C_DI * C_ED;
        const float* brm  = rmat + (long long)blk * C_DI * C_DI;
        const float* bim  = imat + (long long)blk * C_DI * C_DI;
        const float* brb  = rb + blk * C_DI;
        const float* bib  = ib + blk * C_DI;
        const float* bf2  = fc2 + (long long)blk * C_ED * C_DI;

        // --- mamba ---
        k_ln<<<C_BL, 32, 0, stream>>>(h, bl1g, bl1b, xn);
        gemm(xn, 0, C_ED, bip, 0, C_ED, 0, xz, 0, 2 * C_DI,
             nullptr, nullptr, 0, C_BL, 2 * C_DI, C_ED, 1.f, 0.f, 0, 1);
        k_dwconv<<<nEW, 256, 0, stream>>>(xz, bcw, bcb, u);
        gemm(u, 0, C_DI, bxp, 0, C_DI, 0, dbl, 0, 38,
             nullptr, nullptr, 0, C_BL, 38, C_DI, 1.f, 0.f, 0, 1);
        gemm(dbl, 0, 38, bdtw, 0, 6, 0, dtb, 0, C_DI,
             bdtb, nullptr, 0, C_BL, C_DI, 6, 1.f, 0.f, 2, 1);   // softplus
        k_scan<<<C_B * C_DI, 32, 0, stream>>>(dtb, dbl, u, bAl, bDp, ys);
        k_gatez<<<nEW, 256, 0, stream>>>(ys, xz);
        gemm(ys, 0, C_DI, bop, 0, C_DI, 0, h, 0, C_ED,
             nullptr, h, 0, C_BL, C_ED, C_DI, 1.f, 0.f, 0, 1);   // h += mamba

        // --- FFT-FFN ---
        k_ln<<<C_BL, 32, 0, stream>>>(h, bl2g, bl2b, xn);
        gemm(xn, 0, C_ED, bf1, 0, C_ED, 0, hm, 0, C_DI,
             nullptr, nullptr, 0, C_BL, C_DI, C_ED, BN_SCALE_F, 0.f, 1, 1);
        // forward DFT (batched over b): F = Wf · h
        gemm(dftFr, 0, C_L, hm, (long long)C_L * C_DI, C_DI, 1,
             fr, (long long)C_L * C_DI, C_DI, nullptr, nullptr, 0,
             C_L, C_DI, C_L, 1.f, 0.f, 0, C_B);
        gemm(dftFi, 0, C_L, hm, (long long)C_L * C_DI, C_DI, 1,
             fi, (long long)C_L * C_DI, C_DI, nullptr, nullptr, 0,
             C_L, C_DI, C_L, 1.f, 0.f, 0, C_B);
        k_specmix<<<nEW, 256, 0, stream>>>(fr, fi, brm, bim, brb, bib, rp, ip);
        // inverse DFT real part: g = Ir·rp - Ii·ip
        gemm(dftIr, 0, C_L, rp, (long long)C_L * C_DI, C_DI, 1,
             gbuf, (long long)C_L * C_DI, C_DI, nullptr, nullptr, 0,
             C_L, C_DI, C_L, 1.f, 0.f, 0, C_B);
        gemm(dftIi, 0, C_L, ip, (long long)C_L * C_DI, C_DI, 1,
             gbuf, (long long)C_L * C_DI, C_DI, nullptr, nullptr, 0,
             C_L, C_DI, C_L, -1.f, 1.f, 0, C_B);
        gemm(gbuf, 0, C_DI, bf2, 0, C_DI, 0, h, 0, C_ED,
             nullptr, h, 0, C_BL, C_ED, C_DI, BN_SCALE_F, 0.f, 1, 1); // h += relu(..)
    }

    (void)hipMemcpyAsync(d_out, h, (size_t)C_BL * C_ED * sizeof(float),
                         hipMemcpyDeviceToDevice, stream);
}
